// Gnn_7481833030296
// MI455X (gfx1250) — matrix operand order repaired and checked by reading, not compiled, not run on hardware
//
#include <hip/hip_runtime.h>
#include <hip/hip_bf16.h>
#include <stdint.h>

// ---------------------------------------------------------------------------
// GNN (encode -> 2x message passing -> decode) for MI455X / gfx1250.
// Dense layers: v_wmma_f32_16x16x32_bf16 with double-buffered LDS staged via
// CDNA5 async-to-LDS copies (GLOBAL_LOAD_ASYNC_TO_LDS_B128 / ASYNCcnt).
//
// Assumed d_in layout (setup_inputs() dict insertion order, params nested in
// insertion order, each MLP as [(w1,b1),(w2,b2),(w3,b3)]):
//   0 edges i32 [B,E,2]   1 n_feats f32 [B,N,28]   2 e_feats f32 [B,E,3]
//   3 n_feats_const f32 [B,N,4]
//   4.. 9  n_encod   10..15 e_encod   16..21 e_proc[0]  22..27 e_proc[1]
//  28..33  n_proc[0] 34..39 n_proc[1] 40..45 decod
// Output: f32 [B,N,28].
// ---------------------------------------------------------------------------

typedef __bf16 bf16_t;
typedef __attribute__((ext_vector_type(16))) __bf16 v16bf;
typedef __attribute__((ext_vector_type(8)))  __bf16 v8bf;
typedef __attribute__((ext_vector_type(8)))  float  v8f;

static constexpr int GTM = 128;   // M tile
static constexpr int GTN = 64;    // N tile
static constexpr int GKT = 32;    // K tile (= WMMA K)
static constexpr int PAD = 8;     // LDS row padding (bf16 elems)

// One 16-byte async copy global -> LDS (ASYNCcnt-tracked, bypasses VGPRs).
__device__ inline void async_copy_b128(uint32_t lds_off, const void* gaddr) {
    asm volatile("global_load_async_to_lds_b128 %0, %1, off"
                 :: "v"(lds_off), "v"((uint64_t)(uintptr_t)gaddr)
                 : "memory");
}
__device__ inline void wait_async0() {
    asm volatile("s_wait_asynccnt 0" ::: "memory");
}

// ---------------------------------------------------------------------------
// Generic bf16 WMMA GEMM:  out[M,Nc] = act( A[M,K] * W[K,Nc] + bias )
// Wt is pre-transposed/padded: Wt[Nc][K] row-major (K multiple of 32).
// AMODE 0: A dense bf16, leading dim lda.
// AMODE 1: A row r = concat( eLat[r], nLat[edges[2r]], nLat[edges[2r+1]] ),
//          K == 384.
// act: 0 none, 1 relu, 2 residual (x + resid)*0.5.
// Out-of-range tile rows/cols are index-CLAMPED (their outputs are never
// stored; D row m depends only on A row m, D col n only on B col n).
// ---------------------------------------------------------------------------
template<int AMODE>
__global__ __launch_bounds__(256)
void gemm_wmma_kernel(const bf16_t* __restrict__ A, int lda,
                      const bf16_t* __restrict__ eLat,
                      const bf16_t* __restrict__ nLat,
                      const int*    __restrict__ edges,
                      const bf16_t* __restrict__ Wt, int K,
                      const float*  __restrict__ bias,
                      int M, int Nc,
                      bf16_t* __restrict__ outB,
                      float*  __restrict__ outF, int ldo,
                      int act,
                      const bf16_t* __restrict__ resid)
{
    __shared__ __align__(16) bf16_t sA[2][GTM][GKT + PAD];
    __shared__ __align__(16) bf16_t sB[2][GTN][GKT + PAD];

    const int tid  = threadIdx.x;
    const int lane = tid & 31;
    const int wave = tid >> 5;          // 0..7
    const int wm   = wave >> 1;         // 0..3 -> 32-row slab
    const int wn   = wave & 1;          // 0..1 -> 32-col slab
    const int half = lane >> 4;         // 0/1
    const int l16  = lane & 15;
    const int tileM = blockIdx.y * GTM;
    const int tileN = blockIdx.x * GTN;

    // ---- per-thread staging coordinates (fixed across K-tiles) ----
    const int ar  = tid >> 1;                 // A tile row 0..127
    const int akh = (tid & 1) * 16;           // A k-offset within tile {0,16}
    int agrow = tileM + ar; if (agrow >= M) agrow = M - 1;   // clamp
    const int bc  = tid >> 2;                 // B tile col 0..63
    const int bkq = (tid & 3) * 8;            // B k-offset within tile {0,8,16,24}
    int bgcol = tileN + bc; if (bgcol >= Nc) bgcol = Nc - 1; // clamp

    // pre-resolve gather base rows for AMODE 1
    const bf16_t* aRowE = nullptr; const bf16_t* aRowS = nullptr; const bf16_t* aRowD = nullptr;
    if (AMODE == 1) {
        aRowE = eLat + (size_t)agrow * 128;
        aRowS = nLat + (size_t)edges[2 * agrow]     * 128;
        aRowD = nLat + (size_t)edges[2 * agrow + 1] * 128;
    }

    // issue all async copies for K-tile k0 into LDS buffer `buf`
    auto issue_tile = [&](int k0, int buf) {
        const int kk = k0 + akh;
        const bf16_t* srcA;
        if (AMODE == 0) {
            srcA = A + (size_t)agrow * lda + kk;
        } else {
            if (kk < 128)      srcA = aRowE + kk;
            else if (kk < 256) srcA = aRowS + (kk - 128);
            else               srcA = aRowD + (kk - 256);
        }
        const uint32_t la = (uint32_t)(uintptr_t)&sA[buf][ar][akh];
        async_copy_b128(la,      srcA);
        async_copy_b128(la + 16, srcA + 8);
        const bf16_t* srcB = Wt + (size_t)bgcol * K + k0 + bkq;
        const uint32_t lb = (uint32_t)(uintptr_t)&sB[buf][bc][bkq];
        async_copy_b128(lb, srcB);
    };

    v8f zf = {};
    v8f acc[2][2];
    acc[0][0] = zf; acc[0][1] = zf; acc[1][0] = zf; acc[1][1] = zf;

    const int nk = K / GKT;
    issue_tile(0, 0);

    for (int kt = 0; kt < nk; ++kt) {
        const int cur = kt & 1;
        // my async writes into buf[cur] are done:
        wait_async0();
        // everyone's writes into buf[cur] are done; everyone also finished
        // reading buf[1-cur] (compute of kt-1 happened before this barrier):
        __syncthreads();
        if (kt + 1 < nk) issue_tile((kt + 1) * GKT, 1 - cur);

        // ---- fragments per ISA 16-bit layouts, 4 WMMAs ----
        v16bf afrag[2], bfrag[2];
        #pragma unroll
        for (int mi = 0; mi < 2; ++mi) {
            const int row = wm * 32 + mi * 16 + l16;
            v8bf a0 = *(const v8bf*)&sA[cur][row][half * 8];       // K base..+7
            v8bf a1 = *(const v8bf*)&sA[cur][row][half * 8 + 16];  // K base+16..+23
            afrag[mi] = __builtin_shufflevector(a0, a1,
                0,1,2,3,4,5,6,7,8,9,10,11,12,13,14,15);
        }
        #pragma unroll
        for (int ni = 0; ni < 2; ++ni) {
            const int col = wn * 32 + ni * 16 + l16;
            v8bf b0 = *(const v8bf*)&sB[cur][col][half * 8];       // K base..+7
            v8bf b1 = *(const v8bf*)&sB[cur][col][half * 8 + 16];  // K base+16..+23
            bfrag[ni] = __builtin_shufflevector(b0, b1,
                0,1,2,3,4,5,6,7,8,9,10,11,12,13,14,15);
        }
        #pragma unroll
        for (int mi = 0; mi < 2; ++mi)
            #pragma unroll
            for (int ni = 0; ni < 2; ++ni)
                acc[mi][ni] = __builtin_amdgcn_wmma_f32_16x16x32_bf16(
                    false, afrag[mi], false, bfrag[ni],
                    (short)0, acc[mi][ni], false, false);
    }

    // ---- epilogue: bias / relu / residual, bf16 or f32 store ----
    #pragma unroll
    for (int mi = 0; mi < 2; ++mi) {
        #pragma unroll
        for (int ni = 0; ni < 2; ++ni) {
            const int col = tileN + wn * 32 + ni * 16 + l16;
            if (col >= Nc) continue;
            const float bv = bias ? bias[col] : 0.0f;
            const int rbase = tileM + wm * 32 + mi * 16 + half * 8;
            #pragma unroll
            for (int j = 0; j < 8; ++j) {
                const int row = rbase + j;
                if (row >= M) continue;
                float v = acc[mi][ni][j] + bv;
                if (act == 1)      v = v > 0.0f ? v : 0.0f;
                else if (act == 2) v = (v + (float)resid[(size_t)row * ldo + col]) * 0.5f;
                if (outF) outF[(size_t)row * ldo + col] = v;
                else      outB[(size_t)row * ldo + col] = (bf16_t)v;
            }
        }
    }
}

// ---------------------------------------------------------------------------
// Small helper kernels
// ---------------------------------------------------------------------------

// f32 W[K][Nc] -> bf16 Wt[Nc][Kpad] (transpose + zero-pad K)
__global__ void convert_weight_kernel(const float* __restrict__ W,
                                      bf16_t* __restrict__ Wt,
                                      int K, int Nc, int Kpad)
{
    int idx = blockIdx.x * blockDim.x + threadIdx.x;
    if (idx >= Nc * Kpad) return;
    int n = idx / Kpad, k = idx - n * Kpad;
    float v = (k < K) ? W[(size_t)k * Nc + n] : 0.0f;
    Wt[idx] = (bf16_t)v;
}

// concat(n_feats[28], n_feats_const[4]) -> bf16 [N,32]
__global__ void pack_node_in_kernel(const float* __restrict__ nf,
                                    const float* __restrict__ nc,
                                    bf16_t* __restrict__ out, int n)
{
    int idx = blockIdx.x * blockDim.x + threadIdx.x;
    if (idx >= n * 32) return;
    int r = idx >> 5, c = idx & 31;
    float v = (c < 28) ? nf[(size_t)r * 28 + c] : nc[(size_t)r * 4 + (c - 28)];
    out[idx] = (bf16_t)v;
}

// e_feats[3] zero-padded -> bf16 [E,32]
__global__ void pack_edge_in_kernel(const float* __restrict__ ef,
                                    bf16_t* __restrict__ out, int e)
{
    int idx = blockIdx.x * blockDim.x + threadIdx.x;
    if (idx >= e * 32) return;
    int r = idx >> 5, c = idx & 31;
    float v = (c < 3) ? ef[(size_t)r * 3 + c] : 0.0f;
    out[idx] = (bf16_t)v;
}

// concat(n_lat bf16[128], es f32[128]) -> bf16 [N,256]
__global__ void pack_nproc_in_kernel(const bf16_t* __restrict__ nlat,
                                     const float* __restrict__ es,
                                     bf16_t* __restrict__ out, int n)
{
    int idx = blockIdx.x * blockDim.x + threadIdx.x;
    if (idx >= n * 256) return;
    int r = idx >> 8, c = idx & 255;
    out[idx] = (c < 128) ? nlat[(size_t)r * 128 + c]
                         : (bf16_t)es[(size_t)r * 128 + (c - 128)];
}

// segment_sum: es[edges[2e]] += e_lat[e]  (f32 atomics; 4 cols/thread)
__global__ void scatter_add_kernel(const bf16_t* __restrict__ e_lat,
                                   const int* __restrict__ edges,
                                   float* __restrict__ es, int E)
{
    int idx = blockIdx.x * blockDim.x + threadIdx.x;
    int e = idx >> 5;
    if (e >= E) return;
    int c4 = (idx & 31) * 4;
    int n = edges[2 * e];
    #pragma unroll
    for (int j = 0; j < 4; ++j) {
        float v = (float)e_lat[(size_t)e * 128 + c4 + j];
        __hip_atomic_fetch_add(&es[(size_t)n * 128 + c4 + j], v,
                               __ATOMIC_RELAXED, __HIP_MEMORY_SCOPE_AGENT);
    }
}

// ---------------------------------------------------------------------------
extern "C" void kernel_launch(void* const* d_in, const int* in_sizes, int n_in,
                              void* d_out, int out_size, void* d_ws, size_t ws_size,
                              hipStream_t stream)
{
    constexpr int B = 2, N = 10000, E = 160000;
    constexpr int OUTC = 28;
    constexpr int CE = 32768;                 // edge-row chunk for hidden buffers

    const int*   edges_all = (const int*)d_in[0];
    const float* n_feats   = (const float*)d_in[1];
    const float* e_feats   = (const float*)d_in[2];
    const float* n_const   = (const float*)d_in[3];
    auto P = [&](int i) { return (const float*)d_in[i]; };

    // ---- workspace carve (256B aligned slabs) ----
    char* base = (char*)d_ws;
    size_t off = 0;
    auto carve = [&](size_t bytes) -> char* {
        off = (off + 255) & ~(size_t)255;
        char* p = base + off;
        off += bytes;
        return p;
    };
    auto cb = [&](size_t elems) { return (bf16_t*)carve(elems * sizeof(bf16_t)); };

    // weights (bf16, transposed, K padded to mult of 32)
    bf16_t *wt_ne[3], *wt_ee[3], *wt_ep[2][3], *wt_np[2][3], *wt_de[3];
    wt_ne[0] = cb(32  * 256); wt_ne[1] = cb(256 * 256); wt_ne[2] = cb(256 * 128);
    wt_ee[0] = cb(32  * 256); wt_ee[1] = cb(256 * 256); wt_ee[2] = cb(256 * 128);
    for (int i = 0; i < 2; ++i) {
        wt_ep[i][0] = cb(384 * 256); wt_ep[i][1] = cb(256 * 256); wt_ep[i][2] = cb(256 * 128);
    }
    for (int i = 0; i < 2; ++i) {
        wt_np[i][0] = cb(256 * 256); wt_np[i][1] = cb(256 * 256); wt_np[i][2] = cb(256 * 128);
    }
    wt_de[0] = cb(128 * 256); wt_de[1] = cb(256 * 256); wt_de[2] = cb(256 * 28);

    // activations
    bf16_t* nodeIn32 = cb((size_t)N * 32);
    bf16_t* nH1      = cb((size_t)N * 256);
    bf16_t* nH2      = cb((size_t)N * 256);
    bf16_t* nIn256   = cb((size_t)N * 256);
    bf16_t* n_lat    = cb((size_t)N * 128);
    bf16_t* edgeA32  = cb((size_t)E * 32);
    bf16_t* eH1      = cb((size_t)CE * 256);
    bf16_t* eH2      = cb((size_t)CE * 256);
    bf16_t* e_lat    = cb((size_t)E * 128);
    float*  es       = (float*)carve((size_t)N * 128 * sizeof(float));
    (void)ws_size; (void)n_in; (void)in_sizes; (void)out_size;

    // ---- convert all 21 weight matrices once ----
    struct WS { int idx, K, Nc, Kpad; bf16_t* dst; };
    const WS specs[] = {
        { 4, 32, 256,  32, wt_ne[0]}, { 6,256,256,256, wt_ne[1]}, { 8,256,128,256, wt_ne[2]},
        {10,  3, 256,  32, wt_ee[0]}, {12,256,256,256, wt_ee[1]}, {14,256,128,256, wt_ee[2]},
        {16,384, 256, 384, wt_ep[0][0]}, {18,256,256,256, wt_ep[0][1]}, {20,256,128,256, wt_ep[0][2]},
        {22,384, 256, 384, wt_ep[1][0]}, {24,256,256,256, wt_ep[1][1]}, {26,256,128,256, wt_ep[1][2]},
        {28,256, 256, 256, wt_np[0][0]}, {30,256,256,256, wt_np[0][1]}, {32,256,128,256, wt_np[0][2]},
        {34,256, 256, 256, wt_np[1][0]}, {36,256,256,256, wt_np[1][1]}, {38,256,128,256, wt_np[1][2]},
        {40,128, 256, 128, wt_de[0]},   {42,256,256,256, wt_de[1]},   {44,256, 28,256, wt_de[2]},
    };
    for (const WS& s : specs) {
        int cnt = s.Nc * s.Kpad;
        convert_weight_kernel<<<(cnt + 255) / 256, 256, 0, stream>>>(
            P(s.idx), s.dst, s.K, s.Nc, s.Kpad);
    }

    // ---- GEMM launch helper ----
    auto gemm = [&](int amode, const bf16_t* A, int lda,
                    const bf16_t* eL, const int* ed,
                    const bf16_t* Wt, int K, const float* bias,
                    int M, int Nc, bf16_t* outB, float* outF, int ldo,
                    int act, const bf16_t* resid) {
        dim3 grid((Nc + GTN - 1) / GTN, (M + GTM - 1) / GTM);
        if (amode == 0)
            gemm_wmma_kernel<0><<<grid, 256, 0, stream>>>(
                A, lda, nullptr, nullptr, nullptr, Wt, K, bias, M, Nc,
                outB, outF, ldo, act, resid);
        else
            gemm_wmma_kernel<1><<<grid, 256, 0, stream>>>(
                nullptr, 0, eL, n_lat, ed, Wt, K, bias, M, Nc,
                outB, outF, ldo, act, resid);
    };

    for (int b = 0; b < B; ++b) {
        const int*   edg = edges_all + (size_t)b * E * 2;
        const float* nf  = n_feats + (size_t)b * N * 28;
        const float* nc  = n_const + (size_t)b * N * 4;
        const float* ef  = e_feats + (size_t)b * E * 3;
        float* outp = (float*)d_out + (size_t)b * N * OUTC;

        pack_node_in_kernel<<<(N * 32 + 255) / 256, 256, 0, stream>>>(nf, nc, nodeIn32, N);
        pack_edge_in_kernel<<<(E * 32 + 255) / 256, 256, 0, stream>>>(ef, edgeA32, E);

        // node encoder: 32 -> 256 -> 256 -> 128
        gemm(0, nodeIn32, 32, nullptr, nullptr, wt_ne[0], 32,  P(5), N, 256, nH1, nullptr, 256, 1, nullptr);
        gemm(0, nH1, 256, nullptr, nullptr, wt_ne[1], 256, P(7), N, 256, nH2, nullptr, 256, 1, nullptr);
        gemm(0, nH2, 256, nullptr, nullptr, wt_ne[2], 256, P(9), N, 128, n_lat, nullptr, 128, 0, nullptr);

        // edge encoder: 3(pad32) -> 256 -> 256 -> 128, chunked hidden
        for (int e0 = 0; e0 < E; e0 += CE) {
            int m = (E - e0) < CE ? (E - e0) : CE;
            gemm(0, edgeA32 + (size_t)e0 * 32, 32, nullptr, nullptr, wt_ee[0], 32,  P(11), m, 256, eH1, nullptr, 256, 1, nullptr);
            gemm(0, eH1, 256, nullptr, nullptr, wt_ee[1], 256, P(13), m, 256, eH2, nullptr, 256, 1, nullptr);
            gemm(0, eH2, 256, nullptr, nullptr, wt_ee[2], 256, P(15), m, 128, e_lat + (size_t)e0 * 128, nullptr, 128, 0, nullptr);
        }

        // message-passing rounds
        for (int p = 0; p < 2; ++p) {
            const int epB = 16 + p * 6;   // e_proc[p] base idx
            const int npB = 28 + p * 6;   // n_proc[p] base idx
            for (int e0 = 0; e0 < E; e0 += CE) {
                int m = (E - e0) < CE ? (E - e0) : CE;
                bf16_t* eLc = e_lat + (size_t)e0 * 128;
                const int* edc = edg + (size_t)e0 * 2;
                // [e | n_src | n_dst] (384) -> 256 -> 256 -> 128, residual *0.5
                gemm(1, nullptr, 0, eLc, edc, wt_ep[p][0], 384, P(epB + 1), m, 256, eH1, nullptr, 256, 1, nullptr);
                gemm(0, eH1, 256, nullptr, nullptr, wt_ep[p][1], 256, P(epB + 3), m, 256, eH2, nullptr, 256, 1, nullptr);
                gemm(0, eH2, 256, nullptr, nullptr, wt_ep[p][2], 256, P(epB + 5), m, 128, eLc, nullptr, 128, 2, eLc);
            }
            // segment_sum over src nodes
            hipMemsetAsync(es, 0, (size_t)N * 128 * sizeof(float), stream);
            scatter_add_kernel<<<(E * 32 + 255) / 256, 256, 0, stream>>>(e_lat, edg, es, E);
            // node update: [n | es] (256) -> 256 -> 256 -> 128, residual *0.5
            pack_nproc_in_kernel<<<(N * 256 + 255) / 256, 256, 0, stream>>>(n_lat, es, nIn256, N);
            gemm(0, nIn256, 256, nullptr, nullptr, wt_np[p][0], 256, P(npB + 1), N, 256, nH1, nullptr, 256, 1, nullptr);
            gemm(0, nH1, 256, nullptr, nullptr, wt_np[p][1], 256, P(npB + 3), N, 256, nH2, nullptr, 256, 1, nullptr);
            gemm(0, nH2, 256, nullptr, nullptr, wt_np[p][2], 256, P(npB + 5), N, 128, n_lat, nullptr, 128, 2, n_lat);
        }

        // decoder: 128 -> 256 -> 256 -> 28 (f32 out)
        gemm(0, n_lat, 128, nullptr, nullptr, wt_de[0], 128, P(41), N, 256, nH1, nullptr, 256, 1, nullptr);
        gemm(0, nH1, 256, nullptr, nullptr, wt_de[1], 256, P(43), N, 256, nH2, nullptr, 256, 1, nullptr);
        gemm(0, nH2, 256, nullptr, nullptr, wt_de[2], 256, P(45), N, OUTC, nullptr, outp, OUTC, 0, nullptr);
    }
}
